// HGTConv_82832739271042
// MI455X (gfx1250) — matrix operand order repaired and checked by reading, not compiled, not run on hardware
//
#include <hip/hip_runtime.h>
#include <hip/hip_bf16.h>
#include <math.h>

// ---------------------------------------------------------------------------
// Types for CDNA5 WMMA (wave32): 16x16x32 bf16 -> f32
// ---------------------------------------------------------------------------
typedef __bf16 bf16;
typedef __attribute__((ext_vector_type(16))) __bf16 v16bf;
typedef __attribute__((ext_vector_type(8)))  __bf16 v8bf;
typedef __attribute__((ext_vector_type(8)))  float  v8f;

union V16 { v16bf v; v8bf h[2]; };

// Order-preserving float<->uint map so segment-max can use global_atomic_max_u32.
__device__ __forceinline__ unsigned enc_f32(float f) {
    unsigned u = __float_as_uint(f);
    return u ^ ((unsigned)(-(int)(u >> 31)) | 0x80000000u);
}
__device__ __forceinline__ float dec_f32(unsigned e) {
    unsigned m = (e >> 31) ? 0x80000000u : 0xFFFFFFFFu;
    return __uint_as_float(e ^ m);
}
#define ENC_NEG_INF 0x007FFFFFu  // enc_f32(-inf)

// XOR swizzle (16B granules) to spread LDS banks while keeping b128 alignment.
__device__ __forceinline__ int swz(int r, int k) {
    return ((((k >> 3) ^ (r & 7)) << 3) | (k & 7));
}

// ---------------------------------------------------------------------------
// GEMM: C[M,128] = A[M,KDIM] @ W[KDIM,128] + bias.  KDIM in {64,128}.
// Block = 256 threads (8 waves), each block does a 128-row slab.
// Wave w computes rows [w*16, w*16+16) x all 128 cols.
// Per K-step: batch all 18 ds_load_b128 (A-frag + 8 B-frags), one batched
// dscnt wait, then 8 back-to-back v_wmma (XDL co-execution friendly).
// __launch_bounds__(256, 1): lift the occupancy-driven VGPR cap so all 8
// B-frags stay live (acc 64 + B 64 + A 8 + addressing ~ 150 VGPRs); the 8
// waves of one block already fill the WGP, so occupancy loss is moot.
// ---------------------------------------------------------------------------
template <int KDIM>
__launch_bounds__(256, 1)
__global__ void gemm_bias_kernel(const float* __restrict__ A,
                                 const float* __restrict__ W,
                                 const float* __restrict__ bias,
                                 float* __restrict__ C, int M) {
    __shared__ __align__(16) bf16 Alds[128][128]; // [row][k]   (swizzled)
    __shared__ __align__(16) bf16 Wlds[128][128]; // [col][k]   (transposed, swizzled)

    const int tid = threadIdx.x;
    const int rowBase = blockIdx.x * 128;

    // Stage W transposed (bf16).  W is tiny (<=64KB f32) and L2-resident.
    for (int idx = tid; idx < KDIM * 128; idx += 256) {
        int k = idx >> 7;        // W row-major [KDIM][128]
        int c = idx & 127;
        Wlds[c][swz(c, k)] = (bf16)W[idx];
    }
    // Stage A slab (bf16), zero-pad rows past M.
    for (int idx = tid; idx < 128 * KDIM; idx += 256) {
        int r = idx / KDIM;
        int k = idx - r * KDIM;
        int row = rowBase + r;
        Alds[r][swz(r, k)] = (row < M) ? (bf16)A[(size_t)row * KDIM + k] : (bf16)0.0f;
    }
    __syncthreads();

    const int wave = tid >> 5;
    const int lane = tid & 31;
    const int hh = lane >> 4;   // lane half
    const int n  = lane & 15;   // N within tile / M for A-frag
    const int mrow = wave * 16 + n;

    v8f acc[8] = {};

#pragma unroll
    for (int kk = 0; kk < KDIM; kk += 32) {
        // ---- batched loads for this K-step ------------------------------
        // A-frag (16x32): lane half hh -> K groups {kk+8h..+7, kk+16+8h..+7}
        V16 a;
        a.h[0] = *(const v8bf*)&Alds[mrow][swz(mrow, kk + 8 * hh)];
        a.h[1] = *(const v8bf*)&Alds[mrow][swz(mrow, kk + 16 + 8 * hh)];
        // B-frags (32x16): lane = col, lane half hh -> K groups {kk+8h..+7, kk+16+8h..+7}
        V16 b[8];
#pragma unroll
        for (int t = 0; t < 8; ++t) {
            const int col = t * 16 + n;
            b[t].h[0] = *(const v8bf*)&Wlds[col][swz(col, kk + 8 * hh)];
            b[t].h[1] = *(const v8bf*)&Wlds[col][swz(col, kk + 16 + 8 * hh)];
        }
        // ---- 8 back-to-back WMMAs --------------------------------------
#pragma unroll
        for (int t = 0; t < 8; ++t) {
            acc[t] = __builtin_amdgcn_wmma_f32_16x16x32_bf16(
                false, a.v, false, b[t].v, (short)0, acc[t], false, false);
        }
    }

    // C/D layout: VGPR r -> M = 8*hh + r, N = n.
#pragma unroll
    for (int t = 0; t < 8; ++t) {
        const int col = t * 16 + n;
        const float bv = bias[col];
#pragma unroll
        for (int r = 0; r < 8; ++r) {
            int row = rowBase + wave * 16 + 8 * hh + r;
            if (row < M) C[(size_t)row * 128 + col] = acc[t][r] + bv;
        }
    }
}

// ---------------------------------------------------------------------------
// Fill (init) kernel: 16B per lane per iteration.
// ---------------------------------------------------------------------------
__global__ void fill_u32x4(uint4* __restrict__ p, unsigned v, long n4) {
    long i = (long)blockIdx.x * blockDim.x + threadIdx.x;
    if (i < n4) p[i] = make_uint4(v, v, v, v);
}

// ---------------------------------------------------------------------------
// Scores: wave per edge; 4 channels/lane; head = lane>>2 (4 lanes per head).
// score[e,h] = (Q[dst,h,:] . (K[src,h,:] + EW[e,h,:])) / 4
// Also atomic segment-max (monotone-encoded u32).
// ---------------------------------------------------------------------------
__launch_bounds__(256)
__global__ void score_kernel(const int* __restrict__ ei,
                             const float* __restrict__ Kn,
                             const float* __restrict__ Qn,
                             const float* __restrict__ EW,
                             float* __restrict__ scores,
                             unsigned* __restrict__ segmax, int E) {
    const int e = blockIdx.x * 8 + (threadIdx.x >> 5);
    if (e >= E) return;
    const int lane = threadIdx.x & 31;
    const int c0 = lane * 4;
    const int src = ei[e], dst = ei[E + e];

    float4 kv = *(const float4*)(Kn + (size_t)src * 128 + c0);
    float4 qv = *(const float4*)(Qn + (size_t)dst * 128 + c0);
    float4 ev = *(const float4*)(EW + (size_t)e   * 128 + c0);

    float s = qv.x * (kv.x + ev.x) + qv.y * (kv.y + ev.y) +
              qv.z * (kv.z + ev.z) + qv.w * (kv.w + ev.w);
    s += __shfl_xor(s, 1, 32);
    s += __shfl_xor(s, 2, 32);
    s *= 0.25f;  // 1/sqrt(HEAD_DIM=16)

    if ((lane & 3) == 0) {
        const int h = lane >> 2;
        scores[(size_t)e * 8 + h] = s;
        atomicMax(&segmax[(size_t)dst * 8 + h], enc_f32(s));
    }
}

// ---------------------------------------------------------------------------
// ex = exp(score - segmax[dst]);  segsum[dst] += ex  (in-place over scores)
// ---------------------------------------------------------------------------
__launch_bounds__(256)
__global__ void expsum_kernel(const int* __restrict__ ei,
                              const unsigned* __restrict__ segmax,
                              float* __restrict__ scores,
                              float* __restrict__ segsum, int E) {
    long i = (long)blockIdx.x * 256 + threadIdx.x;
    if (i >= (long)E * 8) return;
    int e = (int)(i >> 3), h = (int)(i & 7);
    int dst = ei[E + e];
    float m = dec_f32(segmax[(size_t)dst * 8 + h]);
    if (!__builtin_isfinite(m)) m = 0.0f;
    float ex = __expf(scores[i] - m);
    scores[i] = ex;
    atomicAdd(&segsum[(size_t)dst * 8 + h], ex);
}

// ---------------------------------------------------------------------------
// Scatter: msg[dst] += V[src] * attn.  Wave per edge; msg is L2-resident so
// the f32 atomics resolve in L2 (global_atomic_add_f32).
// ---------------------------------------------------------------------------
__launch_bounds__(256)
__global__ void scatter_kernel(const int* __restrict__ ei,
                               const float* __restrict__ Vn,
                               const float* __restrict__ ex,
                               const float* __restrict__ segsum,
                               float* __restrict__ msg, int E) {
    const int e = blockIdx.x * 8 + (threadIdx.x >> 5);
    if (e >= E) return;
    const int lane = threadIdx.x & 31;
    const int c0 = lane * 4;
    const int h  = lane >> 2;
    const int src = ei[e], dst = ei[E + e];

    float attn = ex[(size_t)e * 8 + h] /
                 (segsum[(size_t)dst * 8 + h] + 1e-16f);
    float4 v = *(const float4*)(Vn + (size_t)src * 128 + c0);
    float* mp = msg + (size_t)dst * 128 + c0;
    atomicAdd(mp + 0, v.x * attn);
    atomicAdd(mp + 1, v.y * attn);
    atomicAdd(mp + 2, v.z * attn);
    atomicAdd(mp + 3, v.w * attn);
}

// ---------------------------------------------------------------------------
// out = leaky_relu(LN(x + t), 0.2).  Wave per row (128 ch, 4/lane).
// ---------------------------------------------------------------------------
__launch_bounds__(256)
__global__ void ln_kernel(const float* __restrict__ x,
                          const float* __restrict__ t,
                          const float* __restrict__ g,
                          const float* __restrict__ beta,
                          float* __restrict__ out, int N) {
    const int row = blockIdx.x * 8 + (threadIdx.x >> 5);
    if (row >= N) return;
    const int lane = threadIdx.x & 31;
    const int c0 = lane * 4;

    float4 xv = *(const float4*)(x + (size_t)row * 128 + c0);
    float4 tv = *(const float4*)(t + (size_t)row * 128 + c0);
    float y[4] = {xv.x + tv.x, xv.y + tv.y, xv.z + tv.z, xv.w + tv.w};

    float s = y[0] + y[1] + y[2] + y[3];
#pragma unroll
    for (int m = 1; m < 32; m <<= 1) s += __shfl_xor(s, m, 32);
    const float mu = s * (1.0f / 128.0f);

    float vs = 0.0f;
#pragma unroll
    for (int j = 0; j < 4; ++j) { float d = y[j] - mu; vs += d * d; }
#pragma unroll
    for (int m = 1; m < 32; m <<= 1) vs += __shfl_xor(vs, m, 32);
    const float rstd = rsqrtf(vs * (1.0f / 128.0f) + 1e-5f);

    float4 gv = *(const float4*)(g + c0);
    float4 bv = *(const float4*)(beta + c0);
    float4 o;
    float z;
    z = (y[0] - mu) * rstd * gv.x + bv.x; o.x = (z > 0.0f) ? z : 0.2f * z;
    z = (y[1] - mu) * rstd * gv.y + bv.y; o.y = (z > 0.0f) ? z : 0.2f * z;
    z = (y[2] - mu) * rstd * gv.z + bv.z; o.z = (z > 0.0f) ? z : 0.2f * z;
    z = (y[3] - mu) * rstd * gv.w + bv.w; o.w = (z > 0.0f) ? z : 0.2f * z;
    *(float4*)(out + (size_t)row * 128 + c0) = o;
}

// ---------------------------------------------------------------------------
// Launcher
// ---------------------------------------------------------------------------
static inline int cdiv(long a, long b) { return (int)((a + b - 1) / b); }

extern "C" void kernel_launch(void* const* d_in, const int* in_sizes, int n_in,
                              void* d_out, int out_size, void* d_ws, size_t ws_size,
                              hipStream_t stream) {
    const float* x_gene = (const float*)d_in[0];
    const float* x_prot = (const float*)d_in[1];
    const int*   ei_ab  = (const int*)d_in[2];
    const int*   ei_ba  = (const int*)d_in[3];
    const float* ee_ab  = (const float*)d_in[4];
    const float* ee_ba  = (const float*)d_in[5];
    const float* Wk_gene = (const float*)d_in[6];  const float* bk_gene = (const float*)d_in[7];
    const float* Wv_gene = (const float*)d_in[8];  const float* bv_gene = (const float*)d_in[9];
    const float* Wq_gene = (const float*)d_in[10]; const float* bq_gene = (const float*)d_in[11];
    const float* Wk_prot = (const float*)d_in[12]; const float* bk_prot = (const float*)d_in[13];
    const float* Wv_prot = (const float*)d_in[14]; const float* bv_prot = (const float*)d_in[15];
    const float* Wq_prot = (const float*)d_in[16]; const float* bq_prot = (const float*)d_in[17];
    const float* We_ab   = (const float*)d_in[18]; const float* be_ab   = (const float*)d_in[19];
    const float* We_ba   = (const float*)d_in[20]; const float* be_ba   = (const float*)d_in[21];
    const float* W_out   = (const float*)d_in[22]; const float* b_out   = (const float*)d_in[23];
    const float* g_gene  = (const float*)d_in[24]; const float* beta_gene = (const float*)d_in[25];
    const float* g_prot  = (const float*)d_in[26]; const float* beta_prot = (const float*)d_in[27];

    const int Ng = in_sizes[0] / 128;
    const int Np = in_sizes[1] / 128;
    const int E  = in_sizes[2] / 2;
    const size_t nmax = (size_t)(Ng > Np ? Ng : Np);

    // Workspace carve-up (floats)
    float*    Kbuf     = (float*)d_ws;
    float*    Vbuf     = Kbuf + nmax * 128;
    float*    Qbuf     = Vbuf + nmax * 128;
    float*    EW       = Qbuf + nmax * 128;           // [E,128]
    float*    scores   = EW + (size_t)E * 128;        // [E,8] (reused as ex)
    unsigned* segmax   = (unsigned*)(scores + (size_t)E * 8); // [nmax,8]
    float*    segsum   = (float*)(segmax + nmax * 8); // [nmax,8]
    float*    msg_gene = segsum + nmax * 8;           // [Ng,128]
    float*    msg_prot = msg_gene + nmax * 128;       // [Np,128]

    const dim3 blk(256);

    // ================= relation (gene --ab--> protein): dst = prot ==========
    fill_u32x4<<<cdiv((long)Np * 2, 256), blk, 0, stream>>>((uint4*)segmax, ENC_NEG_INF, (long)Np * 2);
    fill_u32x4<<<cdiv((long)Np * 2, 256), blk, 0, stream>>>((uint4*)segsum, 0u, (long)Np * 2);
    fill_u32x4<<<cdiv((long)Np * 32, 256), blk, 0, stream>>>((uint4*)msg_prot, 0u, (long)Np * 32);

    gemm_bias_kernel<128><<<cdiv(Ng, 128), blk, 0, stream>>>(x_gene, Wk_gene, bk_gene, Kbuf, Ng);
    gemm_bias_kernel<128><<<cdiv(Ng, 128), blk, 0, stream>>>(x_gene, Wv_gene, bv_gene, Vbuf, Ng);
    gemm_bias_kernel<128><<<cdiv(Np, 128), blk, 0, stream>>>(x_prot, Wq_prot, bq_prot, Qbuf, Np);
    gemm_bias_kernel<64><<<cdiv(E, 128), blk, 0, stream>>>(ee_ab, We_ab, be_ab, EW, E);

    score_kernel<<<cdiv(E, 8), blk, 0, stream>>>(ei_ab, Kbuf, Qbuf, EW, scores, segmax, E);
    expsum_kernel<<<cdiv((long)E * 8, 256), blk, 0, stream>>>(ei_ab, segmax, scores, segsum, E);
    scatter_kernel<<<cdiv(E, 8), blk, 0, stream>>>(ei_ab, Vbuf, scores, segsum, msg_prot, E);

    // ================= relation (prot --ba--> gene): dst = gene =============
    fill_u32x4<<<cdiv((long)Ng * 2, 256), blk, 0, stream>>>((uint4*)segmax, ENC_NEG_INF, (long)Ng * 2);
    fill_u32x4<<<cdiv((long)Ng * 2, 256), blk, 0, stream>>>((uint4*)segsum, 0u, (long)Ng * 2);
    fill_u32x4<<<cdiv((long)Ng * 32, 256), blk, 0, stream>>>((uint4*)msg_gene, 0u, (long)Ng * 32);

    gemm_bias_kernel<128><<<cdiv(Np, 128), blk, 0, stream>>>(x_prot, Wk_prot, bk_prot, Kbuf, Np);
    gemm_bias_kernel<128><<<cdiv(Np, 128), blk, 0, stream>>>(x_prot, Wv_prot, bv_prot, Vbuf, Np);
    gemm_bias_kernel<128><<<cdiv(Ng, 128), blk, 0, stream>>>(x_gene, Wq_gene, bq_gene, Qbuf, Ng);
    gemm_bias_kernel<64><<<cdiv(E, 128), blk, 0, stream>>>(ee_ba, We_ba, be_ba, EW, E);

    score_kernel<<<cdiv(E, 8), blk, 0, stream>>>(ei_ba, Kbuf, Qbuf, EW, scores, segmax, E);
    expsum_kernel<<<cdiv((long)E * 8, 256), blk, 0, stream>>>(ei_ba, segmax, scores, segsum, E);
    scatter_kernel<<<cdiv(E, 8), blk, 0, stream>>>(ei_ba, Vbuf, scores, segsum, msg_gene, E);

    // ================= output projection + residual + LN + leaky ReLU ======
    // K/V buffers are dead now; reuse as GEMM outputs.
    gemm_bias_kernel<128><<<cdiv(Ng, 128), blk, 0, stream>>>(msg_gene, W_out, b_out, Kbuf, Ng);
    gemm_bias_kernel<128><<<cdiv(Np, 128), blk, 0, stream>>>(msg_prot, W_out, b_out, Vbuf, Np);

    float* out = (float*)d_out;
    ln_kernel<<<cdiv(Ng, 8), blk, 0, stream>>>(x_gene, Kbuf, g_gene, beta_gene, out, Ng);
    ln_kernel<<<cdiv(Np, 8), blk, 0, stream>>>(x_prot, Vbuf, g_prot, beta_prot, out + (size_t)Ng * 128, Np);
}
